// GNNStack_90022514524777
// MI455X (gfx1250) — matrix operand order repaired and checked by reading, not compiled, not run on hardware
//
#include <hip/hip_runtime.h>
#include <hip/hip_bf16.h>
#include <math.h>

typedef __attribute__((ext_vector_type(16))) _Float16 v16h;
typedef __attribute__((ext_vector_type(8)))  _Float16 v8h;
typedef __attribute__((ext_vector_type(8)))  float    v8f;

#define WAVE 32

// ---------------------------------------------------------------------------
// Convert fp32 [rowsSrc, K] -> zero-padded f16 [rowsDst, Kp]
// ---------------------------------------------------------------------------
__global__ void cvt_pad_f16(const float* __restrict__ src, _Float16* __restrict__ dst,
                            int rowsDst, int rowsSrc, int K, int Kp)
{
    long i = (long)blockIdx.x * blockDim.x + threadIdx.x;
    long total = (long)rowsDst * Kp;
    if (i >= total) return;
    int row = (int)(i / Kp);
    int col = (int)(i % Kp);
    float v = (row < rowsSrc && col < K) ? src[(size_t)row * K + col] : 0.0f;
    dst[i] = (_Float16)v;
}

// ---------------------------------------------------------------------------
// C[M,NoutStore] = A[M,KP](f16) @ Wh[Noutp,KP](f16)^T + bias, optional ReLU.
// One wave per 16x16 tile. KP is a compile-time multiple of 32, Wh is padded
// to Noutp = tilesN*16 rows -> NO bounds checks in the K loop; per K-step the
// wave issues 4x global_load_b128 + 1x v_wmma_f32_16x16x32_f16, fully
// unrolled (KP/32 WMMAs back-to-back).
// ---------------------------------------------------------------------------
template <int KP>
__global__ void gemm_wmma_f16(const _Float16* __restrict__ A,
                              const _Float16* __restrict__ Wh,
                              const float* __restrict__ bias,
                              float* __restrict__ C,
                              int M, int NoutStore, int tilesN, int relu)
{
    int waveId = (blockIdx.x * blockDim.x + threadIdx.x) / WAVE;
    int lane   = threadIdx.x & (WAVE - 1);
    int tileM  = waveId / tilesN;
    int tileN  = waveId % tilesN;
    if (tileM * 16 >= M) return;
    int m0 = tileM * 16, n0 = tileN * 16;

    int m  = lane & 15;     // A row / B column within tile
    int hi = lane >> 4;     // half-wave selector

    const _Float16* Arow = A  + (size_t)(m0 + m) * KP;
    const _Float16* Wrow = Wh + (size_t)(n0 + m) * KP;   // padded: always valid

    v8f acc = {};
    #pragma unroll
    for (int k0 = 0; k0 < KP; k0 += 32) {
        // A 16x32 f16 fragment: halves [k0+8*hi, +8) then [k0+16+8*hi, +8)
        v8h alo = *(const v8h*)(Arow + k0 + 8 * hi);
        v8h ahi = *(const v8h*)(Arow + k0 + 16 + 8 * hi);
        // B 32x16 f16 fragment: halves [k0+8*hi, +8) then [k0+16+8*hi, +8)
        v8h blo = *(const v8h*)(Wrow + k0 + 8 * hi);
        v8h bhi = *(const v8h*)(Wrow + k0 + 16 + 8 * hi);
        v16h a = __builtin_shufflevector(alo, ahi, 0,1,2,3,4,5,6,7,8,9,10,11,12,13,14,15);
        v16h b = __builtin_shufflevector(blo, bhi, 0,1,2,3,4,5,6,7,8,9,10,11,12,13,14,15);
        acc = __builtin_amdgcn_wmma_f32_16x16x32_f16(false, a, false, b,
                                                     (short)0, acc, false, false);
    }
    int ncol = n0 + m;
    if (ncol < NoutStore) {
        float bv = bias ? bias[ncol] : 0.0f;
        #pragma unroll
        for (int v = 0; v < 8; ++v) {        // C/D: row = v + 8*hi, col = lane&15
            int   row = m0 + v + 8 * hi;
            float val = acc[v] + bv;
            if (relu) val = fmaxf(val, 0.0f);
            C[(size_t)row * NoutStore + ncol] = val;
        }
    }
}

// ---------------------------------------------------------------------------
// Per-(node, head) attention dot products: one wave, 128 channels, shuffle red.
// ---------------------------------------------------------------------------
__global__ void alpha_kernel(const float* __restrict__ h,
                             const float* __restrict__ attl,
                             const float* __restrict__ attr,
                             float* __restrict__ aL, float* __restrict__ aR, int N)
{
    int waveId = (blockIdx.x * blockDim.x + threadIdx.x) / WAVE;
    int lane   = threadIdx.x & 31;
    int node = waveId >> 1, head = waveId & 1;
    if (node >= N) return;
    const float* hv = h + (size_t)node * 256 + head * 128;
    const float* al = attl + head * 128;
    const float* ar = attr + head * 128;
    float sl = 0.f, sr = 0.f;
    #pragma unroll
    for (int i = 0; i < 4; ++i) {
        int c = lane + 32 * i;
        float x = hv[c];
        sl += x * al[c];
        sr += x * ar[c];
    }
    #pragma unroll
    for (int off = 16; off > 0; off >>= 1) {
        sl += __shfl_xor(sl, off, 32);
        sr += __shfl_xor(sr, off, 32);
    }
    if (lane == 0) { aL[node * 2 + head] = sl; aR[node * 2 + head] = sr; }
}

__global__ void fill_kernel(float* __restrict__ p, float v, long n) {
    long i = (long)blockIdx.x * blockDim.x + threadIdx.x;
    if (i < n) p[i] = v;
}

__device__ inline void atomicMaxF(float* addr, float val) {
    if (val >= 0.0f) atomicMax((int*)addr, __float_as_int(val));
    else             atomicMin((unsigned int*)addr, __float_as_uint(val));
}

// e = leaky_relu(aL[src]+aR[dst]); segment max over dst (atomic).
__global__ void edge_score(const int* __restrict__ src, const int* __restrict__ dst,
                           const float* __restrict__ aL, const float* __restrict__ aR,
                           float* __restrict__ esc, float* __restrict__ emax, int E)
{
    int e = blockIdx.x * blockDim.x + threadIdx.x;
    if (e >= E) return;
    int s = src[e], d = dst[e];
    #pragma unroll
    for (int h = 0; h < 2; ++h) {
        float v = aL[s * 2 + h] + aR[d * 2 + h];
        v = (v > 0.f) ? v : 0.2f * v;
        esc[(size_t)e * 2 + h] = v;
        atomicMaxF(&emax[d * 2 + h], v);
    }
}

// ee = exp(e - emax[dst]); segment sum over dst (atomic).
__global__ void edge_expsum(const int* __restrict__ dst, float* __restrict__ esc,
                            const float* __restrict__ emax, float* __restrict__ denom, int E)
{
    int e = blockIdx.x * blockDim.x + threadIdx.x;
    if (e >= E) return;
    int d = dst[e];
    #pragma unroll
    for (int h = 0; h < 2; ++h) {
        float ee = __expf(esc[(size_t)e * 2 + h] - emax[d * 2 + h]);
        esc[(size_t)e * 2 + h] = ee;
        atomicAdd(&denom[d * 2 + h], ee);
    }
}

// One wave per edge: out[dst] += a * h[src] across 256 channels.
// h/out (51 MB each) are L2-resident (192 MB L2) -> gather/scatter stays on-chip.
__global__ void aggregate(const int* __restrict__ src, const int* __restrict__ dst,
                          const float* __restrict__ esc, const float* __restrict__ denom,
                          const float* __restrict__ h, float* __restrict__ out, int E)
{
    int waveId = (blockIdx.x * blockDim.x + threadIdx.x) / WAVE;
    int lane   = threadIdx.x & 31;
    if (waveId >= E) return;
    int s = src[waveId], d = dst[waveId];
    float a0 = esc[(size_t)waveId * 2 + 0] / denom[d * 2 + 0];
    float a1 = esc[(size_t)waveId * 2 + 1] / denom[d * 2 + 1];
    const float* hs = h + (size_t)s * 256;
    float*       od = out + (size_t)d * 256;
    #pragma unroll
    for (int i = 0; i < 8; ++i) {
        int c = lane + 32 * i;
        float coef = (c < 128) ? a0 : a1;
        atomicAdd(&od[c], coef * hs[c]);
    }
}

__global__ void relu_kernel(float* __restrict__ p, long n) {
    long i = (long)blockIdx.x * blockDim.x + threadIdx.x;
    if (i < n) p[i] = fmaxf(p[i], 0.f);
}

// In-place log-softmax, one wave per row of C<=64 columns.
__global__ void log_softmax_inplace(float* __restrict__ x, int N, int C)
{
    int waveId = (blockIdx.x * blockDim.x + threadIdx.x) / WAVE;
    int lane   = threadIdx.x & 31;
    if (waveId >= N) return;
    float* row = x + (size_t)waveId * C;
    float v0 = (lane < C)      ? row[lane]      : -INFINITY;
    float v1 = (lane + 32 < C) ? row[lane + 32] : -INFINITY;
    float m = fmaxf(v0, v1);
    #pragma unroll
    for (int off = 16; off > 0; off >>= 1) m = fmaxf(m, __shfl_xor(m, off, 32));
    float s = 0.f;
    if (lane < C)      s += __expf(v0 - m);
    if (lane + 32 < C) s += __expf(v1 - m);
    #pragma unroll
    for (int off = 16; off > 0; off >>= 1) s += __shfl_xor(s, off, 32);
    float lse = m + __logf(s);
    if (lane < C)      row[lane]      = v0 - lse;
    if (lane + 32 < C) row[lane + 32] = v1 - lse;
}

// ---------------------------------------------------------------------------
extern "C" void kernel_launch(void* const* d_in, const int* in_sizes, int n_in,
                              void* d_out, int out_size, void* d_ws, size_t ws_size,
                              hipStream_t stream)
{
    (void)n_in; (void)out_size; (void)ws_size;
    const int F_IN = 100, HID = 128, H = 2, OUT = 47, CH = H * HID; // 256
    const int FP = 128;    // F_IN padded to multiple of 32
    const int OUTP = 48;   // OUT padded to multiple of 16

    const float* x     = (const float*)d_in[0];
    const int*   ei    = (const int*)  d_in[1];
    const float* W0    = (const float*)d_in[2];
    const float* b0    = (const float*)d_in[3];
    const float* attl0 = (const float*)d_in[4];
    const float* attr0 = (const float*)d_in[5];
    const float* W1    = (const float*)d_in[6];
    const float* b1    = (const float*)d_in[7];
    const float* attl1 = (const float*)d_in[8];
    const float* attr1 = (const float*)d_in[9];
    const float* Wp1   = (const float*)d_in[10];
    const float* bp1   = (const float*)d_in[11];
    const float* Wp2   = (const float*)d_in[12];
    const float* bp2   = (const float*)d_in[13];

    const int N = in_sizes[0] / F_IN;
    const int E = in_sizes[1] / 2;
    const int* src = ei;
    const int* dst = ei + E;

    // ---- workspace carving (16-byte aligned chunks) ----
    char* base = (char*)d_ws;
    size_t off = 0;
    auto carve = [&](size_t bytes) -> char* {
        off = (off + 15) & ~(size_t)15;
        char* p = base + off;
        off += bytes;
        return p;
    };
    float* hbuf  = (float*)carve((size_t)N * CH * 4);   // linear projection h (f32)
    float* agg   = (float*)carve((size_t)N * CH * 4);   // aggregated layer output
    float* aL    = (float*)carve((size_t)N * H * 4);
    float* aR    = (float*)carve((size_t)N * H * 4);
    float* emax  = (float*)carve((size_t)N * H * 4);
    float* denom = (float*)carve((size_t)N * H * 4);
    float* esc   = (float*)carve((size_t)E * H * 4);
    float* hp    = (float*)carve((size_t)N * HID * 4);  // post_mp hidden (f32)
    _Float16* xh   = (_Float16*)carve((size_t)N * FP * 2);    // padded f16 inputs
    _Float16* ah   = (_Float16*)carve((size_t)N * CH * 2);    // f16 activations
    _Float16* hph  = (_Float16*)carve((size_t)N * HID * 2);
    _Float16* W0h  = (_Float16*)carve((size_t)CH * FP * 2);
    _Float16* W1h  = (_Float16*)carve((size_t)CH * CH * 2);
    _Float16* Wp1h = (_Float16*)carve((size_t)HID * CH * 2);
    _Float16* Wp2h = (_Float16*)carve((size_t)OUTP * HID * 2);
    float* logits = (float*)d_out;

    auto cvt = [&](const float* s, _Float16* d, int rD, int rS, int K, int Kp) {
        long total = (long)rD * Kp;
        cvt_pad_f16<<<(unsigned)((total + 255) / 256), 256, 0, stream>>>(s, d, rD, rS, K, Kp);
    };

    // weight conversion (tiny, once per launch)
    cvt(W0,  W0h,  CH,   CH,  F_IN, FP);
    cvt(W1,  W1h,  CH,   CH,  CH,   CH);
    cvt(Wp1, Wp1h, HID,  HID, CH,   CH);
    cvt(Wp2, Wp2h, OUTP, OUT, HID,  HID);

    auto gemm128 = [&](const _Float16* A, const _Float16* Wh, const float* bias,
                       float* C, int NoutStore, int Noutp, int relu) {
        int tilesN = Noutp / 16;
        long waves = (long)((N + 15) / 16) * tilesN;
        gemm_wmma_f16<128><<<(unsigned)((waves + 3) / 4), 128, 0, stream>>>(
            A, Wh, bias, C, N, NoutStore, tilesN, relu);
    };
    auto gemm256 = [&](const _Float16* A, const _Float16* Wh, const float* bias,
                       float* C, int NoutStore, int Noutp, int relu) {
        int tilesN = Noutp / 16;
        long waves = (long)((N + 15) / 16) * tilesN;
        gemm_wmma_f16<256><<<(unsigned)((waves + 3) / 4), 128, 0, stream>>>(
            A, Wh, bias, C, N, NoutStore, tilesN, relu);
    };

    auto attention = [&](const float* al, const float* ar) {
        long waves = (long)N * H;
        alpha_kernel<<<(unsigned)((waves + 7) / 8), 256, 0, stream>>>(hbuf, al, ar, aL, aR, N);
        fill_kernel<<<(N * H + 255) / 256, 256, 0, stream>>>(emax, -INFINITY, (long)N * H);
        fill_kernel<<<(N * H + 255) / 256, 256, 0, stream>>>(denom, 0.f, (long)N * H);
        fill_kernel<<<(unsigned)(((long)N * CH + 255) / 256), 256, 0, stream>>>(
            agg, 0.f, (long)N * CH);
        edge_score<<<(E + 255) / 256, 256, 0, stream>>>(src, dst, aL, aR, esc, emax, E);
        edge_expsum<<<(E + 255) / 256, 256, 0, stream>>>(dst, esc, emax, denom, E);
        aggregate<<<(unsigned)(((long)E + 7) / 8), 256, 0, stream>>>(
            src, dst, esc, denom, hbuf, agg, E);
        relu_kernel<<<(unsigned)(((long)N * CH + 255) / 256), 256, 0, stream>>>(
            agg, (long)N * CH);
    };

    // ---- GAT layer 0 ----
    cvt(x, xh, N, N, F_IN, FP);
    gemm128(xh, W0h, b0, hbuf, CH, CH, 0);
    attention(attl0, attr0);

    // ---- GAT layer 1 ----
    cvt(agg, ah, N, N, CH, CH);              // reads agg before attention() zeroes it
    gemm256(ah, W1h, b1, hbuf, CH, CH, 0);
    attention(attl1, attr1);

    // ---- post_mp ----
    cvt(agg, ah, N, N, CH, CH);
    gemm256(ah, Wp1h, bp1, hp, HID, HID, 0);
    cvt(hp, hph, N, N, HID, HID);
    gemm128(hph, Wp2h, bp2, logits, OUT, OUTP, 0);
    log_softmax_inplace<<<(unsigned)(((long)N + 7) / 8), 256, 0, stream>>>(logits, N, OUT);
}
